// DummyAttention_47974784697228
// MI455X (gfx1250) — matrix operand order repaired and checked by reading, not compiled, not run on hardware
//
#include <hip/hip_runtime.h>
#include <hip/hip_bf16.h>

// ---------------- problem constants ----------------
#define BN 4
#define SN 2048
#define DN 1024
#define HN 16
#define DH 64
#define ML 1536

typedef __attribute__((ext_vector_type(16))) __bf16 v16bf;
typedef __attribute__((ext_vector_type(8)))  __bf16 v8bf;
typedef __attribute__((ext_vector_type(8)))  float  v8f;

#define SHUF16(lo, hi) __builtin_shufflevector(lo, hi, 0,1,2,3,4,5,6,7,8,9,10,11,12,13,14,15)
#define WMMA_BF16(a, b, c) \
  __builtin_amdgcn_wmma_f32_16x16x32_bf16(false, (a), false, (b), (short)0, (c), false, false)

// ---------------- fp32 -> bf16 conversion ----------------
__global__ __launch_bounds__(256) void cvt_bf16_kernel(const float* __restrict__ in,
                                                       __bf16* __restrict__ out, int n) {
  int t = blockIdx.x * blockDim.x + threadIdx.x;
  int stride = gridDim.x * blockDim.x;
  for (int i = t; i < n; i += stride) out[i] = (__bf16)in[i];
}

// ---------------- ragged trim: kept indices + lengths ----------------
__global__ void build_idx_kernel(const unsigned char* __restrict__ skip,
                                 int* __restrict__ idx, int* __restrict__ len) {
  int b = blockIdx.x;
  if (threadIdx.x != 0) return;
  int c = 0;
  for (int s = 0; s < SN; ++s) {
    if (skip[b * SN + s]) {
      if (c < ML) idx[b * ML + c] = s;
      ++c;
    }
  }
  int L = c < ML ? c : ML;
  for (int j = L; j < ML; ++j) idx[b * ML + j] = 0;
  len[b] = L;
}

// ---------------- BF16 WMMA GEMM: C[M,N] = A[M,K] * W[N,K]^T ----------------
// wave computes 32(M) x 64(N); block = 8 waves stacked over M -> 256 x 64 per block.
__global__ __launch_bounds__(256)
void gemm_bf16_kernel(const __bf16* __restrict__ A, const __bf16* __restrict__ W,
                      __bf16* __restrict__ C, int M, int N, int K) {
  const int lane = threadIdx.x & 31;
  const int w    = threadIdx.x >> 5;
  const int half = lane >> 4;
  const int n16  = lane & 15;
  const int rowbase = blockIdx.x * 256 + w * 32;
  const int colbase = blockIdx.y * 64;

  v8f acc[2][4];
  for (int i = 0; i < 2; ++i)
    for (int j = 0; j < 4; ++j)
      for (int e = 0; e < 8; ++e) acc[i][j][e] = 0.0f;

  const __bf16* arow0 = A + (size_t)(rowbase + n16) * K;
  const __bf16* arow1 = A + (size_t)(rowbase + 16 + n16) * K;

  for (int kb = 0; kb < K; kb += 32) {
    __builtin_prefetch(arow0 + kb + 128, 0, 1);
    __builtin_prefetch(arow1 + kb + 128, 0, 1);
    v16bf a0, a1;
    {
      v8bf lo = *(const v8bf*)(arow0 + kb + 16 * half);
      v8bf hi = *(const v8bf*)(arow0 + kb + 8 + 16 * half);
      a0 = SHUF16(lo, hi);
      lo = *(const v8bf*)(arow1 + kb + 16 * half);
      hi = *(const v8bf*)(arow1 + kb + 8 + 16 * half);
      a1 = SHUF16(lo, hi);
    }
    for (int nt = 0; nt < 4; ++nt) {
      const __bf16* brow = W + (size_t)(colbase + nt * 16 + n16) * K + kb + 16 * half;
      v16bf bf = *(const v16bf*)brow;
      acc[0][nt] = WMMA_BF16(a0, bf, acc[0][nt]);
      acc[1][nt] = WMMA_BF16(a1, bf, acc[1][nt]);
    }
  }
  for (int mi = 0; mi < 2; ++mi)
    for (int nt = 0; nt < 4; ++nt)
      for (int v = 0; v < 8; ++v) {
        int r = rowbase + mi * 16 + v + 8 * half;
        int c = colbase + nt * 16 + n16;
        C[(size_t)r * N + c] = (__bf16)acc[mi][nt][v];
      }
}

// ---------------- output projection GEMM: fp32 out, invalid rows zeroed ----------------
__global__ __launch_bounds__(256)
void gemm_out_kernel(const __bf16* __restrict__ A, const __bf16* __restrict__ W,
                     float* __restrict__ C, const int* __restrict__ len,
                     int M, int N, int K) {
  const int lane = threadIdx.x & 31;
  const int w    = threadIdx.x >> 5;
  const int half = lane >> 4;
  const int n16  = lane & 15;
  const int rowbase = blockIdx.x * 256 + w * 32;
  const int colbase = blockIdx.y * 64;

  v8f acc[2][4];
  for (int i = 0; i < 2; ++i)
    for (int j = 0; j < 4; ++j)
      for (int e = 0; e < 8; ++e) acc[i][j][e] = 0.0f;

  const __bf16* arow0 = A + (size_t)(rowbase + n16) * K;
  const __bf16* arow1 = A + (size_t)(rowbase + 16 + n16) * K;

  for (int kb = 0; kb < K; kb += 32) {
    __builtin_prefetch(arow0 + kb + 128, 0, 1);
    v16bf a0, a1;
    {
      v8bf lo = *(const v8bf*)(arow0 + kb + 16 * half);
      v8bf hi = *(const v8bf*)(arow0 + kb + 8 + 16 * half);
      a0 = SHUF16(lo, hi);
      lo = *(const v8bf*)(arow1 + kb + 16 * half);
      hi = *(const v8bf*)(arow1 + kb + 8 + 16 * half);
      a1 = SHUF16(lo, hi);
    }
    for (int nt = 0; nt < 4; ++nt) {
      const __bf16* brow = W + (size_t)(colbase + nt * 16 + n16) * K + kb + 16 * half;
      v16bf bf = *(const v16bf*)brow;
      acc[0][nt] = WMMA_BF16(a0, bf, acc[0][nt]);
      acc[1][nt] = WMMA_BF16(a1, bf, acc[1][nt]);
    }
  }
  for (int mi = 0; mi < 2; ++mi)
    for (int nt = 0; nt < 4; ++nt)
      for (int v = 0; v < 8; ++v) {
        int r = rowbase + mi * 16 + v + 8 * half;
        int b = r / ML, q = r % ML;
        bool ok = q < len[b];
        int c = colbase + nt * 16 + n16;
        C[(size_t)r * N + c] = ok ? acc[mi][nt][v] : 0.0f;
      }
}

// ---------------- RoPE on gathered (trimmed) Q ----------------
__global__ __launch_bounds__(256)
void rope_gather_q_kernel(const __bf16* __restrict__ Qp, const int* __restrict__ idx,
                          const int* __restrict__ len, const float* __restrict__ inv_freq,
                          __bf16* __restrict__ Qt) {
  int t = blockIdx.x * blockDim.x + threadIdx.x;
  const int total = BN * ML * HN * 32;
  if (t >= total) return;
  int i = t & 31;
  int h = (t >> 5) & (HN - 1);
  int rest = t >> 9;
  int l = rest % ML;
  int b = rest / ML;
  int L = len[b];
  int row = idx[b * ML + l];
  float pos = (l < L) ? (float)row : 0.0f;
  const __bf16* src = Qp + ((size_t)(b * SN) + row) * DN + h * DH;
  float x1 = (float)src[i];
  float x2 = (float)src[i + 32];
  float ang = pos * inv_freq[i];
  float c = cosf(ang), s = sinf(ang);
  __bf16* dst = Qt + ((size_t)(b * ML) + l) * DN + h * DH;
  dst[i]      = (__bf16)(x1 * c - x2 * s);
  dst[i + 32] = (__bf16)(x2 * c + x1 * s);
}

// ---------------- RoPE on K (full positions), in place ----------------
__global__ __launch_bounds__(256)
void rope_k_kernel(__bf16* __restrict__ Kp, const float* __restrict__ inv_freq) {
  int t = blockIdx.x * blockDim.x + threadIdx.x;
  const int total = BN * SN * HN * 32;
  if (t >= total) return;
  int i = t & 31;
  int h = (t >> 5) & (HN - 1);
  int rest = t >> 9;
  int s = rest % SN;
  int b = rest / SN;
  __bf16* p = Kp + ((size_t)(b * SN) + s) * DN + h * DH;
  float x1 = (float)p[i];
  float x2 = (float)p[i + 32];
  float ang = (float)s * inv_freq[i];
  float cc = cosf(ang), ss = sinf(ang);
  p[i]      = (__bf16)(x1 * cc - x2 * ss);
  p[i + 32] = (__bf16)(x2 * cc + x1 * ss);
}

// ---------------- V transpose: [b,s,h*64+dh] -> [b,h,dh,s] ----------------
__global__ __launch_bounds__(256)
void transpose_v_kernel(const __bf16* __restrict__ Vp, __bf16* __restrict__ Vt) {
  int t = blockIdx.x * blockDim.x + threadIdx.x;
  const int total = BN * SN * DN;
  if (t >= total) return;
  int e = t & (DN - 1);
  int s = (t >> 10) & (SN - 1);
  int b = t >> 21;
  int h = e >> 6;
  int dh = e & (DH - 1);
  Vt[(((size_t)(b * HN + h)) * DH + dh) * SN + s] = Vp[t];
}

// ---------------- flash attention: 1 wave = 16 queries x full head ----------------
__global__ __launch_bounds__(32)
void attn_kernel(const __bf16* __restrict__ Qt, const __bf16* __restrict__ Kp,
                 const __bf16* __restrict__ Vt, const int* __restrict__ idx,
                 const int* __restrict__ len, __bf16* __restrict__ Obf) {
  __shared__ __bf16 sP[16 * 32];  // P tile bounce: C-layout -> A-layout
  const int lane = threadIdx.x;
  const int half = lane >> 4;
  const int n16  = lane & 15;
  const int qbase = blockIdx.x * 16;
  const int h = blockIdx.y;
  const int b = blockIdx.z;
  const float SCALE = 0.125f;  // 1/sqrt(64)

  const int L = len[b];
  int qp[8];
  int mq = -1;
  for (int v = 0; v < 8; ++v) {
    int q = qbase + v + 8 * half;
    int p = idx[b * ML + q];
    qp[v] = (q < L) ? p : -1;
    mq = max(mq, qp[v]);
  }
  mq = max(mq, __shfl_xor(mq, 1, 32));
  mq = max(mq, __shfl_xor(mq, 2, 32));
  mq = max(mq, __shfl_xor(mq, 4, 32));
  mq = max(mq, __shfl_xor(mq, 8, 32));
  mq = max(mq, __shfl_xor(mq, 16, 32));

  if (mq < 0) {  // whole tile invalid (uniform across wave)
    for (int t4 = 0; t4 < 4; ++t4)
      for (int v = 0; v < 8; ++v) {
        int m = v + 8 * half;
        Obf[((size_t)(b * ML) + qbase + m) * DN + h * DH + 16 * t4 + n16] = (__bf16)0.0f;
      }
    return;
  }
  const int kend = ((mq + 32) / 32) * 32;

  // Q A-fragments over dh (K dim = 64 -> two K=32 fragments)
  const __bf16* qrow = Qt + ((size_t)(b * ML) + qbase + n16) * DN + h * DH;
  v16bf aQ0, aQ1;
  {
    v8bf lo = *(const v8bf*)(qrow + 16 * half);
    v8bf hi = *(const v8bf*)(qrow + 8 + 16 * half);
    aQ0 = SHUF16(lo, hi);
    lo = *(const v8bf*)(qrow + 32 + 16 * half);
    hi = *(const v8bf*)(qrow + 40 + 16 * half);
    aQ1 = SHUF16(lo, hi);
  }

  float rm[8], rl[8];
  v8f o[4];
  for (int v = 0; v < 8; ++v) { rm[v] = -1e30f; rl[v] = 0.0f; }
  for (int t4 = 0; t4 < 4; ++t4)
    for (int e = 0; e < 8; ++e) o[t4][e] = 0.0f;

  const v8f czero = {};

  for (int kb = 0; kb < kend; kb += 32) {
    const __bf16* krow0 = Kp + ((size_t)(b * SN) + kb + n16) * DN + h * DH;
    const __bf16* krow1 = krow0 + (size_t)16 * DN;
    __builtin_prefetch(krow0 + (size_t)32 * DN, 0, 1);

    // score tiles: keys [kb, kb+16) and [kb+16, kb+32)
    v16bf bK;
    v8f ca, cb;
    bK = *(const v16bf*)(krow0 + 16 * half);
    ca = WMMA_BF16(aQ0, bK, czero);
    bK = *(const v16bf*)(krow0 + 32 + 16 * half);
    ca = WMMA_BF16(aQ1, bK, ca);
    bK = *(const v16bf*)(krow1 + 16 * half);
    cb = WMMA_BF16(aQ0, bK, czero);
    bK = *(const v16bf*)(krow1 + 32 + 16 * half);
    cb = WMMA_BF16(aQ1, bK, cb);

    __syncthreads();  // S_NOP for 1-wave WG; WAR fence vs prior LDS reads
    float alpha[8];
    for (int v = 0; v < 8; ++v) {
      float s0 = ca[v] * SCALE;
      float s1 = cb[v] * SCALE;
      int key0 = kb + n16;
      s0 = (key0 <= qp[v]) ? s0 : -1e9f;
      s1 = (key0 + 16 <= qp[v]) ? s1 : -1e9f;
      float t = fmaxf(s0, s1);
      t = fmaxf(t, __shfl_xor(t, 1, 32));
      t = fmaxf(t, __shfl_xor(t, 2, 32));
      t = fmaxf(t, __shfl_xor(t, 4, 32));
      t = fmaxf(t, __shfl_xor(t, 8, 32));
      float mn = fmaxf(rm[v], t);
      float a = __expf(rm[v] - mn);
      float p0 = __expf(s0 - mn);
      float p1 = __expf(s1 - mn);
      float ps = p0 + p1;
      ps += __shfl_xor(ps, 1, 32);
      ps += __shfl_xor(ps, 2, 32);
      ps += __shfl_xor(ps, 4, 32);
      ps += __shfl_xor(ps, 8, 32);
      rl[v] = rl[v] * a + ps;
      rm[v] = mn;
      alpha[v] = a;
      int m = v + 8 * half;
      sP[m * 32 + n16]      = (__bf16)p0;
      sP[m * 32 + 16 + n16] = (__bf16)p1;
    }
    __syncthreads();  // DS stores visible before A-fragment reload

    // P tile as A-fragment (16 rows x 32 keys)
    v16bf aP;
    {
      v8bf lo = *(const v8bf*)&sP[n16 * 32 + 16 * half];
      v8bf hi = *(const v8bf*)&sP[n16 * 32 + 8 + 16 * half];
      aP = SHUF16(lo, hi);
    }

    for (int t4 = 0; t4 < 4; ++t4) {
      for (int v = 0; v < 8; ++v) o[t4][v] *= alpha[v];
      const __bf16* vrow = Vt + (((size_t)(b * HN + h)) * DH + 16 * t4 + n16) * SN + kb;
      v16bf bV = *(const v16bf*)(vrow + 16 * half);
      o[t4] = WMMA_BF16(aP, bV, o[t4]);
    }
  }

  for (int t4 = 0; t4 < 4; ++t4)
    for (int v = 0; v < 8; ++v) {
      int m = v + 8 * half;
      float inv = (rl[v] > 0.0f) ? (1.0f / rl[v]) : 0.0f;
      Obf[((size_t)(b * ML) + qbase + m) * DN + h * DH + 16 * t4 + n16] =
          (__bf16)(o[t4][v] * inv);
    }
}

// ---------------- host launch ----------------
extern "C" void kernel_launch(void* const* d_in, const int* in_sizes, int n_in,
                              void* d_out, int out_size, void* d_ws, size_t ws_size,
                              hipStream_t stream) {
  const float* q_src = (const float*)d_in[0];
  const float* k_src = (const float*)d_in[1];
  const float* v_src = (const float*)d_in[2];
  const float* Wq = (const float*)d_in[3];
  const float* Wk = (const float*)d_in[4];
  const float* Wv = (const float*)d_in[5];
  const float* Wo = (const float*)d_in[6];
  const float* inv_freq = (const float*)d_in[7];
  const unsigned char* skip = (const unsigned char*)d_in[10];
  float* out = (float*)d_out;

  char* p = (char*)d_ws;
  auto take = [&](size_t bytes) -> void* {
    void* r = (void*)p;
    p += (bytes + 255) & ~(size_t)255;
    return r;
  };
  int*    idx = (int*)take((size_t)BN * ML * 4);
  int*    len = (int*)take((size_t)BN * 4);
  __bf16* WqB = (__bf16*)take((size_t)DN * DN * 2);
  __bf16* WkB = (__bf16*)take((size_t)DN * DN * 2);
  __bf16* WvB = (__bf16*)take((size_t)DN * DN * 2);
  __bf16* WoB = (__bf16*)take((size_t)DN * DN * 2);
  __bf16* QsB = (__bf16*)take((size_t)BN * SN * DN * 2);
  __bf16* KsB = (__bf16*)take((size_t)BN * SN * DN * 2);
  __bf16* VsB = (__bf16*)take((size_t)BN * SN * DN * 2);
  __bf16* Qp  = (__bf16*)take((size_t)BN * SN * DN * 2);
  __bf16* Kp  = (__bf16*)take((size_t)BN * SN * DN * 2);
  __bf16* Vp  = (__bf16*)take((size_t)BN * SN * DN * 2);
  __bf16* Qt  = (__bf16*)take((size_t)BN * ML * DN * 2);
  __bf16* Vt  = (__bf16*)take((size_t)BN * HN * DH * SN * 2);
  __bf16* Obf = (__bf16*)take((size_t)BN * ML * DN * 2);

  const int nW = DN * DN;        // 1,048,576
  const int nS = BN * SN * DN;   // 8,388,608
  cvt_bf16_kernel<<<(nW + 255) / 256, 256, 0, stream>>>(Wq, WqB, nW);
  cvt_bf16_kernel<<<(nW + 255) / 256, 256, 0, stream>>>(Wk, WkB, nW);
  cvt_bf16_kernel<<<(nW + 255) / 256, 256, 0, stream>>>(Wv, WvB, nW);
  cvt_bf16_kernel<<<(nW + 255) / 256, 256, 0, stream>>>(Wo, WoB, nW);
  cvt_bf16_kernel<<<(nS + 255) / 256, 256, 0, stream>>>(q_src, QsB, nS);
  cvt_bf16_kernel<<<(nS + 255) / 256, 256, 0, stream>>>(k_src, KsB, nS);
  cvt_bf16_kernel<<<(nS + 255) / 256, 256, 0, stream>>>(v_src, VsB, nS);

  build_idx_kernel<<<BN, 64, 0, stream>>>(skip, idx, len);

  // projections: [B*S, D] x [D, D]^T
  {
    dim3 g((BN * SN) / 256, DN / 64);
    gemm_bf16_kernel<<<g, 256, 0, stream>>>(QsB, WqB, Qp, BN * SN, DN, DN);
    gemm_bf16_kernel<<<g, 256, 0, stream>>>(KsB, WkB, Kp, BN * SN, DN, DN);
    gemm_bf16_kernel<<<g, 256, 0, stream>>>(VsB, WvB, Vp, BN * SN, DN, DN);
  }

  {
    int n = BN * ML * HN * 32;
    rope_gather_q_kernel<<<(n + 255) / 256, 256, 0, stream>>>(Qp, idx, len, inv_freq, Qt);
  }
  {
    int n = BN * SN * HN * 32;
    rope_k_kernel<<<(n + 255) / 256, 256, 0, stream>>>(Kp, inv_freq);
  }
  {
    int n = BN * SN * DN;
    transpose_v_kernel<<<(n + 255) / 256, 256, 0, stream>>>(Vp, Vt);
  }

  {
    dim3 g(ML / 16, HN, BN);
    attn_kernel<<<g, 32, 0, stream>>>(Qt, Kp, Vt, idx, len, Obf);
  }

  // output projection: [B*ML, 1024] x Wo[1024,1024]^T -> fp32, masked
  {
    dim3 g((BN * ML) / 256, DN / 64);
    gemm_out_kernel<<<g, 256, 0, stream>>>(Obf, WoB, out, len, BN * ML, DN, DN);
  }
}
